// LSTM_47218870452857
// MI455X (gfx1250) — compile-verified, matrix operand order checked by reading, not run on hardware
//
#include <hip/hip_runtime.h>
#include <hip/hip_bf16.h>
#include <math.h>

#define H 1024
#define B 2048
#define NL 4
#define BH (B * H)

#define BM 128
#define BN 128
#define BK 32
#define THREADS 256      // 8 waves of 32
#define LDS_STRIDE 40    // 32 halves + 8 pad -> 80B row, 16B aligned

typedef __attribute__((ext_vector_type(16))) _Float16 v16h;
typedef __attribute__((ext_vector_type(8)))  _Float16 v8h;
typedef __attribute__((ext_vector_type(8)))  float    v8f;

union FragU { v16h v; v8h h[2]; };

#if __has_builtin(__builtin_amdgcn_s_wait_asynccnt)
#define WAIT_ASYNC0() __builtin_amdgcn_s_wait_asynccnt(0)
#else
#define WAIT_ASYNC0() asm volatile("s_wait_asynccnt 0" ::: "memory")
#endif

// ---------------------------------------------------------------------------
// Prep: fp32 -> f16 activation conversion + cell copy
// ---------------------------------------------------------------------------
__global__ void lstm_prep(const float* __restrict__ x_in,
                          const float* __restrict__ h_in,
                          const float* __restrict__ c_in,
                          _Float16* __restrict__ xh,
                          _Float16* __restrict__ hh,
                          float* __restrict__ c_ws) {
    int idx = blockIdx.x * blockDim.x + threadIdx.x;
    if (idx < BH) {
        xh[idx]   = (_Float16)x_in[idx];
        hh[idx]   = (_Float16)h_in[idx];
        c_ws[idx] = c_in[idx];
    }
}

// ---------------------------------------------------------------------------
// WMMA GEMM: gates[g] = Xf16 * Wih[g,l] + Hf16 * Whh[g,l]   (f32 accum)
// grid: (M/BM=16, 1024/BN=8, gates=4), block: 256 threads (8 waves)
// ---------------------------------------------------------------------------

// A tile (f16, no conversion): async global->LDS, 2 x b128 per thread.
__device__ __forceinline__ void load_a_tile_async(_Float16* sA,
                                                  const _Float16* __restrict__ A,
                                                  int m0, int k0, int tid) {
    int row = tid >> 1;        // 0..127
    int ch  = tid & 1;         // two 16-half (32B) chunks per row
    const _Float16* src = A + (size_t)(m0 + row) * H + k0 + ch * 16;
    unsigned lds = (unsigned)(size_t)(sA + row * LDS_STRIDE + ch * 16);
    unsigned long long ga = (unsigned long long)(size_t)src;
    asm volatile(
        "global_load_async_to_lds_b128 %0, %2, off\n\t"
        "global_load_async_to_lds_b128 %1, %2, off offset:16"
        :: "v"(lds), "v"(lds + 16u), "v"(ga)
        : "memory");
}

// B tile: 32(K) x 128(N) fp32 row-major -> LDS transposed n-major f16.
__device__ __forceinline__ void load_b_tile(_Float16* sB,
                                            const float* __restrict__ W,
                                            int k0, int n0, int tid) {
    int k   = tid >> 3;         // 0..31
    int nch = tid & 7;          // 0..7 -> 16 cols each
    const float4* src = reinterpret_cast<const float4*>(W + (size_t)(k0 + k) * H + n0 + nch * 16);
    float4 f0 = src[0], f1 = src[1], f2 = src[2], f3 = src[3];
    _Float16 h[16] = {
        (_Float16)f0.x, (_Float16)f0.y, (_Float16)f0.z, (_Float16)f0.w,
        (_Float16)f1.x, (_Float16)f1.y, (_Float16)f1.z, (_Float16)f1.w,
        (_Float16)f2.x, (_Float16)f2.y, (_Float16)f2.z, (_Float16)f2.w,
        (_Float16)f3.x, (_Float16)f3.y, (_Float16)f3.z, (_Float16)f3.w};
#pragma unroll
    for (int i = 0; i < 16; ++i)
        sB[(nch * 16 + i) * LDS_STRIDE + k] = h[i];
}

__device__ __forceinline__ v16h load_frag(const _Float16* s, int r_off, int lo16, int hi) {
    // 16-bit A/B fragment: lane row/col = lane&15; chunks at halves 8*hi and 16+8*hi
    const _Float16* p = s + (r_off + lo16) * LDS_STRIDE + 8 * hi;
    FragU u;
    u.h[0] = *reinterpret_cast<const v8h*>(p);
    u.h[1] = *reinterpret_cast<const v8h*>(p + 16);
    return u.v;
}

__global__ void __launch_bounds__(THREADS)
lstm_gemm(const _Float16* __restrict__ Ax,   // x activations  [B,H] f16
          const _Float16* __restrict__ Ah,   // hidden         [B,H] f16
          const float* __restrict__ w_ih,    // [4,NL,H,H] fp32
          const float* __restrict__ w_hh,    // [4,NL,H,H] fp32
          int layer,
          float* __restrict__ gates) {       // [4,B,H] fp32
    __shared__ _Float16 sA[2][BM * LDS_STRIDE];
    __shared__ _Float16 sB[2][BN * LDS_STRIDE];

    const int tid  = threadIdx.x;
    const int g    = blockIdx.z;
    const int m0   = blockIdx.x * BM;
    const int n0   = blockIdx.y * BN;

    const float* Wih = w_ih + ((size_t)g * NL + layer) * (size_t)H * H;
    const float* Whh = w_hh + ((size_t)g * NL + layer) * (size_t)H * H;

    const int wave = tid >> 5;
    const int lane = tid & 31;
    const int lo16 = lane & 15;
    const int hi   = lane >> 4;
    const int wm   = wave & 3;   // 4 waves in M -> 32 rows each
    const int wn   = wave >> 2;  // 2 waves in N -> 64 cols each

    v8f acc[2][4] = {};

    const int NSTEP = 2 * (H / BK);   // 64: first 32 over (Ax,Wih), then (Ah,Whh)

    // prologue
    load_a_tile_async(sA[0], Ax, m0, 0, tid);
    load_b_tile(sB[0], Wih, 0, n0, tid);
    WAIT_ASYNC0();
    __syncthreads();

    for (int s = 0; s < NSTEP; ++s) {
        const int cur = s & 1;
        const int nxt = cur ^ 1;

        if (s + 1 < NSTEP) {
            const _Float16* An = (s + 1 < NSTEP / 2) ? Ax : Ah;
            const float*    Wn = (s + 1 < NSTEP / 2) ? Wih : Whh;
            const int k0n = ((s + 1) & (NSTEP / 2 - 1)) * BK;
            load_a_tile_async(sA[nxt], An, m0, k0n, tid);
            load_b_tile(sB[nxt], Wn, k0n, n0, tid);
        }
        if (s + 2 < NSTEP) {
            const float* Wp = (s + 2 < NSTEP / 2) ? Wih : Whh;
            const int k0p = ((s + 2) & (NSTEP / 2 - 1)) * BK;
            __builtin_prefetch(Wp + (size_t)(k0p + (tid >> 3)) * H + n0 + (tid & 7) * 16, 0, 0);
        }

        v16h af[2], bf[4];
#pragma unroll
        for (int i = 0; i < 2; ++i)
            af[i] = load_frag(sA[cur], wm * 32 + i * 16, lo16, hi);
#pragma unroll
        for (int j = 0; j < 4; ++j)
            bf[j] = load_frag(sB[cur], wn * 64 + j * 16, lo16, hi);

#pragma unroll
        for (int i = 0; i < 2; ++i)
#pragma unroll
            for (int j = 0; j < 4; ++j)
                acc[i][j] = __builtin_amdgcn_wmma_f32_16x16x32_f16(
                    false, af[i], false, bf[j], (short)0, acc[i][j], false, false);

        WAIT_ASYNC0();      // this wave's async A loads into sA[nxt] have landed
        __syncthreads();
    }

    // epilogue: C layout -> lane: N = lane&15, M = v + 8*(lane>>4)
    float* Cg = gates + (size_t)g * BH;
#pragma unroll
    for (int i = 0; i < 2; ++i) {
#pragma unroll
        for (int j = 0; j < 4; ++j) {
            const int mt = m0 + wm * 32 + i * 16 + 8 * hi;
            const int nt = n0 + wn * 64 + j * 16 + lo16;
            float* cp = Cg + (size_t)mt * H + nt;
#pragma unroll
            for (int v = 0; v < 8; ++v)
                cp[(size_t)v * H] = acc[i][j][v];
        }
    }
}

// ---------------------------------------------------------------------------
// Elementwise LSTM update
// ---------------------------------------------------------------------------
__device__ __forceinline__ float sigmoidf_(float x) { return 1.0f / (1.0f + __expf(-x)); }

__global__ void lstm_elem(const float* __restrict__ gates,   // [4,B,H]
                          const float* __restrict__ b_ih,    // [4,NL,1,H]
                          const float* __restrict__ b_hh,
                          int layer,
                          float* __restrict__ c_ws,          // [B,H]
                          _Float16* __restrict__ x_next,     // [B,H] f16
                          float* __restrict__ out,           // [3,B,H] (last layer)
                          int is_last) {
    int idx = blockIdx.x * blockDim.x + threadIdx.x;
    if (idx >= BH) return;
    const int k = idx & (H - 1);

    float gi = gates[0 * (size_t)BH + idx] + b_ih[(0 * NL + layer) * H + k] + b_hh[(0 * NL + layer) * H + k];
    float gf = gates[1 * (size_t)BH + idx] + b_ih[(1 * NL + layer) * H + k] + b_hh[(1 * NL + layer) * H + k];
    float gg = gates[2 * (size_t)BH + idx] + b_ih[(2 * NL + layer) * H + k] + b_hh[(2 * NL + layer) * H + k];
    float go = gates[3 * (size_t)BH + idx] + b_ih[(3 * NL + layer) * H + k] + b_hh[(3 * NL + layer) * H + k];

    float it = sigmoidf_(gi);
    float ft = sigmoidf_(gf);
    float gt = tanhf(gg);
    float ot = sigmoidf_(go);

    float c_new = ft * c_ws[idx] + it * gt;
    float x_new = ot * tanhf(c_new);

    c_ws[idx]   = c_new;
    x_next[idx] = (_Float16)x_new;

    if (is_last) {
        out[idx]                  = ot;
        out[(size_t)BH + idx]     = x_new;
        out[2 * (size_t)BH + idx] = c_new;
    }
}

// ---------------------------------------------------------------------------
extern "C" void kernel_launch(void* const* d_in, const int* in_sizes, int n_in,
                              void* d_out, int out_size, void* d_ws, size_t ws_size,
                              hipStream_t stream) {
    const float* input_in = (const float*)d_in[0];
    const float* hidden   = (const float*)d_in[1];
    const float* cell     = (const float*)d_in[2];
    const float* w_ih     = (const float*)d_in[3];
    const float* w_hh     = (const float*)d_in[4];
    const float* b_ih     = (const float*)d_in[5];
    const float* b_hh     = (const float*)d_in[6];
    float* out = (float*)d_out;

    char* ws = (char*)d_ws;
    float*    gates = (float*)ws;                               // 4*B*H*4  = 32MB
    float*    c_ws  = (float*)(ws + (size_t)4 * BH * 4);        // B*H*4    =  8MB
    _Float16* xh0   = (_Float16*)(ws + (size_t)5 * BH * 4);     // B*H*2    =  4MB
    _Float16* xh1   = (_Float16*)(ws + (size_t)5 * BH * 4 + (size_t)BH * 2);
    _Float16* hh    = (_Float16*)(ws + (size_t)5 * BH * 4 + (size_t)2 * BH * 2);
    _Float16* xbuf[2] = {xh0, xh1};

    const int eb = 256;
    const int eg = (BH + eb - 1) / eb;

    lstm_prep<<<eg, eb, 0, stream>>>(input_in, hidden, cell, xh0, hh, c_ws);

    dim3 ggrid(B / BM, H / BN, 4);
    for (int l = 0; l < NL; ++l) {
        lstm_gemm<<<ggrid, THREADS, 0, stream>>>(xbuf[l & 1], hh, w_ih, w_hh, l, gates);
        lstm_elem<<<eg, eb, 0, stream>>>(gates, b_ih, b_hh, l, c_ws,
                                         xbuf[(l + 1) & 1], out, l == NL - 1);
    }
}
